// Gemma3Attention_2705829397233
// MI455X (gfx1250) — matrix operand order repaired and checked by reading, not compiled, not run on hardware
//
#include <hip/hip_runtime.h>
#include <hip/hip_bf16.h>
#include <math.h>
#include <stdint.h>

// ---------------------------------------------------------------------------
// Gemma3 attention block for MI455X (gfx1250, wave32, WMMA + TDM).
// GEMMs: v_wmma_f32_16x16x32_bf16 (f32 accumulate).
// Attention K/V tiles: double-buffered tensor_load_to_lds (TDM, TENSORcnt),
// DMA of chunk i+1 overlapped with WMMA compute of chunk i.
// ---------------------------------------------------------------------------

typedef __attribute__((ext_vector_type(16))) __bf16 v16bf;
typedef __attribute__((ext_vector_type(8)))  __bf16 v8bf;
typedef __attribute__((ext_vector_type(8)))  float  v8f;
typedef __attribute__((ext_vector_type(4)))  unsigned int u32x4;
typedef __attribute__((ext_vector_type(4)))  int i32x4;
typedef __attribute__((ext_vector_type(8)))  int i32x8;

#define S_LEN   2048
#define HDIM    2560
#define NHQ     8
#define NKV     4
#define HD      256
#define GROUPS  2
#define WINDOW  1024
#define SCALING 0.0625f        // 256^-0.5
#define SOFTCAP 50.0f
#define RMS_EPS 1e-6f
#define QKV_N   4096           // 2048 q + 1024 k + 1024 v columns

#define WMMA_BF16(A, B, C) \
  __builtin_amdgcn_wmma_f32_16x16x32_bf16(false, (A), false, (B), (short)0, (C), false, false)

// A-matrix fragment (16x32 bf16): lane m = lane&15, half = lane>>4.
// Elements 0..7  = K: half*16 + 0..7     (contiguous 16B)
// Elements 8..15 = K: half*16 + 8..15    (contiguous 16B, +16B from first)
__device__ inline v16bf load_a_frag(const __bf16* p) {
  v8bf lo = *(const v8bf*)(p);
  v8bf hi = *(const v8bf*)(p + 8);
  return __builtin_shufflevector(lo, hi, 0,1,2,3,4,5,6,7,8,9,10,11,12,13,14,15);
}

// ---------------------------------------------------------------------------
// TDM: 2-D tile load Global -> LDS via Tensor Data Mover.
// D# packing per CDNA5 ISA 8.3/8.4: group0 = {count|flags, lds_addr,
// global_addr[31:0], global_addr[56:32]|type=2}; group1 carries data_size,
// tensor_dim0/1, tile_dim0/1, tensor_dim0_stride. Groups 2/3 zero (2-D).
// Arity differs between toolchains (bridge doc): clang-22 = 5 args,
// clang-23 = 6 args.
// ---------------------------------------------------------------------------
__device__ inline void tdm_load_2d(unsigned lds_addr, unsigned long long gaddr,
                                   unsigned tensor_d0, unsigned tensor_d1,
                                   unsigned tile_d0, unsigned tile_d1,
                                   unsigned stride_d0) {
  u32x4 g0;
  g0[0] = 1u;                                        // count=1, user mode
  g0[1] = lds_addr;                                  // LDS byte address
  g0[2] = (unsigned)(gaddr & 0xFFFFFFFFu);           // global_addr[31:0]
  g0[3] = (unsigned)((gaddr >> 32) & 0x01FFFFFFu)    // global_addr[56:32]
        | (2u << 30);                                // type = 2 ("image")
  i32x8 g1;
  g1[0] = 0x00010000;                                // wg_mask=0, data_size=2B
  g1[1] = (int)((tensor_d0 & 0xFFFFu) << 16);        // dim0[15:0] -> [63:48]
  g1[2] = (int)((tensor_d0 >> 16) | ((tensor_d1 & 0xFFFFu) << 16));
  g1[3] = (int)((tensor_d1 >> 16) | (tile_d0 << 16)); // tile_dim0 -> [127:112]
  g1[4] = (int)tile_d1;                              // tile_dim1; tile_dim2=0
  g1[5] = (int)stride_d0;                            // dim0_stride[31:0]
  g1[6] = 0;                                         // stride hi / dim1_stride
  g1[7] = 0;
  i32x4 z4 = {};
#if defined(__clang_major__) && (__clang_major__ >= 23)
  i32x8 z8 = {};
  __builtin_amdgcn_tensor_load_to_lds(g0, g1, z4, z4, z8, 0);
#else
  __builtin_amdgcn_tensor_load_to_lds(g0, g1, z4, z4, 0);
#endif
}

// ---------------------------------------------------------------------------
// Elementwise fp32 -> bf16 (n multiple of 256)
// ---------------------------------------------------------------------------
__global__ void f32_to_bf16_kernel(const float* __restrict__ in,
                                   __bf16* __restrict__ out, size_t n) {
  size_t i = (size_t)blockIdx.x * 256 + threadIdx.x;
  if (i < n) out[i] = (__bf16)in[i];
}

// ---------------------------------------------------------------------------
// WMMA GEMM: C[m,n] = sum_k A[m,k] * W[n,k]   (i.e. A @ W^T)
// A: [M,K] bf16 row-major (lda), W: [N,K] bf16 row-major (ldw), C fp32 (ldc).
// Block = 256 threads = 8 waves -> 128x64 tile; wave = 32x32 (2x2 WMMA tiles).
// Requires M%128==0, N%64==0, K%32==0.
// ---------------------------------------------------------------------------
__global__ void gemm_bf16_wmma(const __bf16* __restrict__ A, int lda,
                               const __bf16* __restrict__ W, int ldw,
                               float* __restrict__ C, int ldc, int K) {
  const int lane = threadIdx.x & 31;
  const int wave = threadIdx.x >> 5;
  const int half = lane >> 4;
  const int l16  = lane & 15;
  const int row0 = blockIdx.y * 128 + (wave & 3) * 32;
  const int col0 = blockIdx.x * 64  + (wave >> 2) * 32;

  v8f acc00 = {}, acc01 = {}, acc10 = {}, acc11 = {};
  const __bf16* arow0 = A + (size_t)(row0 + l16) * lda + half * 16;
  const __bf16* arow1 = A + (size_t)(row0 + 16 + l16) * lda + half * 16;
  const __bf16* wrow0 = W + (size_t)(col0 + l16) * ldw + half * 16;
  const __bf16* wrow1 = W + (size_t)(col0 + 16 + l16) * ldw + half * 16;

  for (int kk = 0; kk < K; kk += 32) {
    v16bf a0 = load_a_frag(arow0 + kk);
    v16bf a1 = load_a_frag(arow1 + kk);
    v16bf b0 = *(const v16bf*)(wrow0 + kk);
    v16bf b1 = *(const v16bf*)(wrow1 + kk);
    acc00 = WMMA_BF16(a0, b0, acc00);
    acc01 = WMMA_BF16(a0, b1, acc01);
    acc10 = WMMA_BF16(a1, b0, acc10);
    acc11 = WMMA_BF16(a1, b1, acc11);
  }
#pragma unroll
  for (int v = 0; v < 8; ++v) {
    const int m0 = row0 + v + 8 * half;
    const int n0 = col0 + l16;
    C[(size_t)m0 * ldc + n0]             = acc00[v];
    C[(size_t)m0 * ldc + n0 + 16]        = acc01[v];
    C[(size_t)(m0 + 16) * ldc + n0]      = acc10[v];
    C[(size_t)(m0 + 16) * ldc + n0 + 16] = acc11[v];
  }
}

// ---------------------------------------------------------------------------
// RMSNorm (over head_dim) + RoPE for Q and K, fp32 -> bf16, head-major out.
// grid = (S, NHQ + NKV), block = 256 (== HD)
// ---------------------------------------------------------------------------
__global__ void qk_norm_rope_kernel(const float* __restrict__ qkvf,
                                    const float* __restrict__ cosb,
                                    const float* __restrict__ sinb,
                                    const float* __restrict__ qnw,
                                    const float* __restrict__ knw,
                                    __bf16* __restrict__ qb,
                                    __bf16* __restrict__ kb) {
  __shared__ float red[8];
  __shared__ float nsh[HD];
  const int s = blockIdx.x;
  const int h = blockIdx.y;
  const int d = threadIdx.x;

  const float* w;
  __bf16* outp;
  int col;
  if (h < NHQ) {
    col = h * HD + d;
    w = qnw;
    outp = qb + ((size_t)h * S_LEN + s) * HD + d;
  } else {
    const int kv = h - NHQ;
    col = NHQ * HD + kv * HD + d;
    w = knw;
    outp = kb + ((size_t)kv * S_LEN + s) * HD + d;
  }
  const float x = qkvf[(size_t)s * QKV_N + col];

  float ss = x * x;
#pragma unroll
  for (int m = 1; m <= 16; m <<= 1) ss += __shfl_xor(ss, m, 32);
  if ((threadIdx.x & 31) == 0) red[threadIdx.x >> 5] = ss;
  __syncthreads();
  float tot = 0.f;
#pragma unroll
  for (int i = 0; i < 8; ++i) tot += red[i];
  const float rms = rsqrtf(tot / (float)HD + RMS_EPS);
  const float nv = x * rms * (1.0f + w[d]);
  nsh[d] = nv;
  __syncthreads();
  const float rot = (d < HD / 2) ? -nsh[d + HD / 2] : nsh[d - HD / 2];
  const float val = nv * cosb[(size_t)s * HD + d] + rot * sinb[(size_t)s * HD + d];
  *outp = (__bf16)val;
}

// ---------------------------------------------------------------------------
// V: fp32 [s, 2048 + kv*HD + d] -> bf16 transposed [kv][d][s]
// ---------------------------------------------------------------------------
__global__ void v_transpose_kernel(const float* __restrict__ qkvf,
                                   __bf16* __restrict__ vtb) {
  size_t i = (size_t)blockIdx.x * 256 + threadIdx.x;  // over NKV*S*HD
  const int kv = (int)(i / ((size_t)S_LEN * HD));
  const size_t rem = i % ((size_t)S_LEN * HD);
  const int s = (int)(rem / HD);
  const int d = (int)(rem % HD);
  vtb[((size_t)kv * HD + d) * S_LEN + s] =
      (__bf16)qkvf[(size_t)s * QKV_N + NHQ * HD + kv * HD + d];
}

// ---------------------------------------------------------------------------
// Flash attention with softcap + sliding-window mask.
// grid = (S/128, NHQ), block = 256 (8 waves). Each wave owns 16 query rows.
// Keys are processed in block-uniform chunks of 32. Each chunk's K tile
// ([32 keys][256 d]) and V tile ([256 d][32 keys]) is DMA'd into LDS by the
// Tensor Data Mover, double-buffered: chunk i+1's DMA overlaps chunk i's
// WMMA compute (TENSORcnt in-order completion => wait tensorcnt<=2).
// Q: [head][s][d] bf16, K: [kv][s][d] bf16, Vt: [kv][d][s] bf16,
// O: [s][head*HD + d] bf16.
// ---------------------------------------------------------------------------
__global__ void attn_wmma_kernel(const __bf16* __restrict__ Q,
                                 const __bf16* __restrict__ Kh,
                                 const __bf16* __restrict__ Vt,
                                 __bf16* __restrict__ O) {
  __shared__ __attribute__((aligned(64))) __bf16 ksh[2][32 * HD];   // 2x16KB [key][d]
  __shared__ __attribute__((aligned(64))) __bf16 vsh[2][HD * 32];   // 2x16KB [d][key]
  __shared__ __attribute__((aligned(64))) __bf16 pbuf[8][16 * 32];  // 8KB

  const int lane = threadIdx.x & 31;
  const int wave = threadIdx.x >> 5;
  const int half = lane >> 4;
  const int l16  = lane & 15;
  const int head = blockIdx.y;
  const int kvh  = head / GROUPS;
  const int q0blk = blockIdx.x * 128;
  const int q0   = q0blk + wave * 16;

  // Cache Q fragments for all of head_dim (8 x K=32 steps), lane row = l16.
  const __bf16* qrow = Q + ((size_t)head * S_LEN + (q0 + l16)) * HD + half * 16;
  v16bf qf[8];
#pragma unroll
  for (int t = 0; t < 8; ++t) qf[t] = load_a_frag(qrow + 32 * t);

  v8f acc[16];
#pragma unroll
  for (int i = 0; i < 16; ++i) acc[i] = (v8f){};
  v8f mst, lst;
#pragma unroll
  for (int v = 0; v < 8; ++v) { mst[v] = -3.0e38f; lst[v] = 0.f; }

  const unsigned long long kgbase =
      (unsigned long long)(uintptr_t)(Kh + (size_t)kvh * S_LEN * HD);
  const unsigned long long vgbase =
      (unsigned long long)(uintptr_t)(Vt + (size_t)kvh * HD * S_LEN);
  unsigned ksh_a[2], vsh_a[2];
  ksh_a[0] = (unsigned)(unsigned long long)(uintptr_t)&ksh[0][0];
  ksh_a[1] = (unsigned)(unsigned long long)(uintptr_t)&ksh[1][0];
  vsh_a[0] = (unsigned)(unsigned long long)(uintptr_t)&vsh[0][0];
  vsh_a[1] = (unsigned)(unsigned long long)(uintptr_t)&vsh[1][0];
  __bf16* pb = pbuf[wave];

  // Block-uniform key range (per-row window enforced by the mask below).
  int kmin = q0blk - (WINDOW - 1);
  if (kmin < 0) kmin = 0;
  kmin &= ~31;
  int kmax = q0blk + 127 + WINDOW;      // exclusive
  if (kmax > S_LEN) kmax = S_LEN;
  const int nch = (kmax - kmin + 31) >> 5;

  // Prologue: start DMA for chunk 0 into buffer 0.
  if (wave == 0) {
    tdm_load_2d(ksh_a[0], kgbase + (unsigned long long)kmin * (HD * 2),
                HD, S_LEN, HD, 32, HD);
    tdm_load_2d(vsh_a[0], vgbase + (unsigned long long)kmin * 2,
                S_LEN, HD, 32, HD, S_LEN);
  }

  for (int c = 0; c < nch; ++c) {
    const int kc  = kmin + (c << 5);
    const int cur = c & 1;
    if (wave == 0) {
      if (c + 1 < nch) {
        // Issue chunk c+1 into the other buffer (its previous readers were
        // released by the closing barrier of iteration c-1), then wait until
        // only those 2 in-flight ops remain => chunk c's DMA is complete.
        const unsigned long long kn = (unsigned long long)(kmin + ((c + 1) << 5));
        tdm_load_2d(ksh_a[cur ^ 1], kgbase + kn * (HD * 2),
                    HD, S_LEN, HD, 32, HD);
        tdm_load_2d(vsh_a[cur ^ 1], vgbase + kn * 2,
                    S_LEN, HD, 32, HD, S_LEN);
        __builtin_amdgcn_s_wait_tensorcnt(2);
      } else {
        __builtin_amdgcn_s_wait_tensorcnt(0);
      }
    }
    __syncthreads();                    // publish chunk c tiles to all waves
    const __bf16* ks = ksh[cur];
    const __bf16* vs = vsh[cur];

    // ---- scores: S = Q K^T over d (two 16-key column tiles) ----
    v8f s0 = {}, s1 = {};
    const __bf16* kr0 = ks + (size_t)l16 * HD + half * 16;
    const __bf16* kr1 = ks + (size_t)(16 + l16) * HD + half * 16;
#pragma unroll
    for (int t = 0; t < 8; ++t) {
      v16bf b0 = *(const v16bf*)(kr0 + 32 * t);
      s0 = WMMA_BF16(qf[t], b0, s0);
    }
#pragma unroll
    for (int t = 0; t < 8; ++t) {
      v16bf b1 = *(const v16bf*)(kr1 + 32 * t);
      s1 = WMMA_BF16(qf[t], b1, s1);
    }
    // ---- scale, softcap (tanh), sliding-window mask ----
#pragma unroll
    for (int v = 0; v < 8; ++v) {
      const int qi = q0 + v + 8 * half;
      const int k0 = kc + l16;
      const int k1 = k0 + 16;
      float x0 = SOFTCAP * tanhf(s0[v] * SCALING * (1.0f / SOFTCAP));
      float x1 = SOFTCAP * tanhf(s1[v] * SCALING * (1.0f / SOFTCAP));
      int d0 = qi - k0; d0 = d0 < 0 ? -d0 : d0;
      int d1 = qi - k1; d1 = d1 < 0 ? -d1 : d1;
      if (d0 >= WINDOW) x0 += -1e9f;
      if (d1 >= WINDOW) x1 += -1e9f;
      s0[v] = x0; s1[v] = x1;
    }
    // ---- online softmax: row reductions across the 16-lane half ----
    v8f rmax;
#pragma unroll
    for (int v = 0; v < 8; ++v) rmax[v] = fmaxf(s0[v], s1[v]);
#pragma unroll
    for (int m = 1; m <= 8; m <<= 1)
#pragma unroll
      for (int v = 0; v < 8; ++v) rmax[v] = fmaxf(rmax[v], __shfl_xor(rmax[v], m, 32));

    v8f p0, p1, rsum;
#pragma unroll
    for (int v = 0; v < 8; ++v) {
      const float mnew = fmaxf(mst[v], rmax[v]);
      const float sc = __expf(mst[v] - mnew);
      p0[v] = __expf(s0[v] - mnew);
      p1[v] = __expf(s1[v] - mnew);
      rsum[v] = p0[v] + p1[v];
      lst[v] *= sc;
      mst[v] = mnew;
      s0[v] = sc;  // reuse s0 as rescale factor
    }
#pragma unroll
    for (int m = 1; m <= 8; m <<= 1)
#pragma unroll
      for (int v = 0; v < 8; ++v) rsum[v] += __shfl_xor(rsum[v], m, 32);
#pragma unroll
    for (int v = 0; v < 8; ++v) lst[v] += rsum[v];
#pragma unroll
    for (int nt = 0; nt < 16; ++nt)
#pragma unroll
      for (int v = 0; v < 8; ++v) acc[nt][v] *= s0[v];

    // ---- P: C/D layout -> A layout via per-wave LDS tile ----
#pragma unroll
    for (int v = 0; v < 8; ++v) {
      const int r = v + 8 * half;
      pb[r * 32 + l16]      = (__bf16)p0[v];
      pb[r * 32 + 16 + l16] = (__bf16)p1[v];
    }
    asm volatile("s_wait_dscnt 0" ::: "memory");
    v16bf pf = load_a_frag(pb + l16 * 32 + half * 16);

    // ---- O += P V : 16 output column tiles over head_dim ----
#pragma unroll
    for (int nt = 0; nt < 16; ++nt) {
      v16bf vf = *(const v16bf*)(vs + (size_t)(16 * nt + l16) * 32 + half * 16);
      acc[nt] = WMMA_BF16(pf, vf, acc[nt]);
    }
    __syncthreads();                    // readers done; buffer may be refilled
  }

  // ---- normalize and store bf16 to [s][head*HD + d] ----
#pragma unroll
  for (int nt = 0; nt < 16; ++nt)
#pragma unroll
    for (int v = 0; v < 8; ++v) {
      const int m = q0 + v + 8 * half;
      const int d = 16 * nt + l16;
      O[(size_t)m * (NHQ * HD) + head * HD + d] = (__bf16)(acc[nt][v] / lst[v]);
    }
}

// ---------------------------------------------------------------------------
// Host launcher
// ---------------------------------------------------------------------------
extern "C" void kernel_launch(void* const* d_in, const int* in_sizes, int n_in,
                              void* d_out, int out_size, void* d_ws, size_t ws_size,
                              hipStream_t stream) {
  const float* hs   = (const float*)d_in[0];
  const float* cosb = (const float*)d_in[1];
  const float* sinb = (const float*)d_in[2];
  // d_in[3] = attention_mask (recomputed analytically from WINDOW)
  const float* Wq   = (const float*)d_in[4];
  const float* Wk   = (const float*)d_in[5];
  const float* Wv   = (const float*)d_in[6];
  const float* Wo   = (const float*)d_in[7];
  const float* qnw  = (const float*)d_in[8];
  const float* knw  = (const float*)d_in[9];
  float* out = (float*)d_out;
  (void)in_sizes; (void)n_in; (void)out_size; (void)ws_size;

  unsigned char* base = (unsigned char*)d_ws;
  size_t off = 0;
#define CARVE(T, name, nbytes)                       \
  T name = (T)(base + off);                          \
  off += (((size_t)(nbytes)) + 255) & ~(size_t)255;

  CARVE(__bf16*, hsb,  (size_t)S_LEN * HDIM * 2)
  CARVE(__bf16*, wqb,  (size_t)NHQ * HD * HDIM * 2)
  CARVE(__bf16*, wkb,  (size_t)NKV * HD * HDIM * 2)
  CARVE(__bf16*, wvb,  (size_t)NKV * HD * HDIM * 2)
  CARVE(__bf16*, wob,  (size_t)HDIM * NHQ * HD * 2)
  CARVE(float*,  qkvf, (size_t)S_LEN * QKV_N * 4)
  CARVE(__bf16*, qb,   (size_t)NHQ * S_LEN * HD * 2)
  CARVE(__bf16*, kbuf, (size_t)NKV * S_LEN * HD * 2)
  CARVE(__bf16*, vtb,  (size_t)NKV * HD * S_LEN * 2)
  CARVE(__bf16*, aob,  (size_t)S_LEN * NHQ * HD * 2)
#undef CARVE

  // 1) fp32 -> bf16 conversions (all counts are multiples of 256)
  const size_t n_hs = (size_t)S_LEN * HDIM;
  const size_t n_wq = (size_t)NHQ * HD * HDIM;
  const size_t n_wk = (size_t)NKV * HD * HDIM;
  const size_t n_wo = (size_t)HDIM * NHQ * HD;
  f32_to_bf16_kernel<<<(unsigned)(n_hs / 256), 256, 0, stream>>>(hs, hsb, n_hs);
  f32_to_bf16_kernel<<<(unsigned)(n_wq / 256), 256, 0, stream>>>(Wq, wqb, n_wq);
  f32_to_bf16_kernel<<<(unsigned)(n_wk / 256), 256, 0, stream>>>(Wk, wkb, n_wk);
  f32_to_bf16_kernel<<<(unsigned)(n_wk / 256), 256, 0, stream>>>(Wv, wvb, n_wk);
  f32_to_bf16_kernel<<<(unsigned)(n_wo / 256), 256, 0, stream>>>(Wo, wob, n_wo);

  // 2) QKV projections into qkvf [S, 4096] (Q | K | V)
  gemm_bf16_wmma<<<dim3((NHQ * HD) / 64, S_LEN / 128), 256, 0, stream>>>(
      hsb, HDIM, wqb, HDIM, qkvf, QKV_N, HDIM);
  gemm_bf16_wmma<<<dim3((NKV * HD) / 64, S_LEN / 128), 256, 0, stream>>>(
      hsb, HDIM, wkb, HDIM, qkvf + NHQ * HD, QKV_N, HDIM);
  gemm_bf16_wmma<<<dim3((NKV * HD) / 64, S_LEN / 128), 256, 0, stream>>>(
      hsb, HDIM, wvb, HDIM, qkvf + NHQ * HD + NKV * HD, QKV_N, HDIM);

  // 3) RMSNorm + RoPE (Q,K) and V transpose, emitting bf16
  qk_norm_rope_kernel<<<dim3(S_LEN, NHQ + NKV), HD, 0, stream>>>(
      qkvf, cosb, sinb, qnw, knw, qb, kbuf);
  v_transpose_kernel<<<(unsigned)(((size_t)NKV * S_LEN * HD) / 256), 256, 0, stream>>>(
      qkvf, vtb);

  // 4) Flash attention: double-buffered TDM K/V tiles, softcap + window
  attn_wmma_kernel<<<dim3(S_LEN / 128, NHQ), 256, 0, stream>>>(qb, kbuf, vtb, aob);

  // 5) Output projection: out = attn_out @ Wo^T  -> fp32 [S, HDIM]
  gemm_bf16_wmma<<<dim3(HDIM / 64, S_LEN / 128), 256, 0, stream>>>(
      aob, NHQ * HD, wob, NHQ * HD, out, HDIM, NHQ * HD);
}
